// BidirLSTMLayer_6811818132103
// MI455X (gfx1250) — matrix operand order repaired and checked by reading, not compiled, not run on hardware
//
#include <hip/hip_runtime.h>

// ---------------------------------------------------------------------------
// Bidirectional LSTM for MI455X (gfx1250, wave32, WMMA, async-LDS staging).
//  Phase 0: convert W_ih / W_hh to bf16, reset grid-barrier state.
//  Phase 1: x_proj = x @ W_ih^T + (b_ih + b_hh)  via v_wmma_f32_16x16x32_bf16,
//           W_ih block staged with global_load_async_to_lds_b128.
//  Phase 2: persistent recurrent kernel, 64 WGs (2 dirs x 32 column slices):
//           W_hh slice async-staged into LDS (64KB, loaded once), c-state in
//           registers, h exchanged via L2-resident bf16 ping-pong + grid barrier.
// Workspace use: ~546 MB (x_proj fp32 dominates).
// ---------------------------------------------------------------------------

typedef __attribute__((ext_vector_type(16))) __bf16 v16bf;
typedef __attribute__((ext_vector_type(8)))  __bf16 v8bf;
typedef __attribute__((ext_vector_type(8)))  float  v8f;

#define T_STEPS 512
#define BATCH   64
#define HID     512
#define G4      2048   // 4*HID

static __device__ __forceinline__ float sigm(float x) {
    return 1.0f / (1.0f + __expf(-x));
}

static __device__ __forceinline__ v16bf cat8(v8bf lo, v8bf hi) {
    return __builtin_shufflevector(lo, hi, 0,1,2,3,4,5,6,7,8,9,10,11,12,13,14,15);
}

// Async 16-byte copy global -> LDS (ASYNCcnt-tracked, no VGPR data movement).
// ldsoff is the byte offset inside the block's (dynamic) LDS allocation.
static __device__ __forceinline__ void async_copy_b128(unsigned ldsoff,
                                                       const void* gsrc) {
    asm volatile("global_load_async_to_lds_b128 %0, %1, off"
                 :: "v"(ldsoff), "v"(gsrc) : "memory");
}
static __device__ __forceinline__ void async_wait0() {
    asm volatile("s_wait_asynccnt 0" ::: "memory");
}

// ---------------------------------------------------------------------------
// Phase 0: fp32 -> bf16 weight conversion + barrier reset
// ---------------------------------------------------------------------------
__global__ __launch_bounds__(256) void cvt_weights_kernel(
    const float* __restrict__ Wihf, const float* __restrict__ Whhf,
    const float* __restrict__ Wihb, const float* __restrict__ Whhb,
    __bf16* __restrict__ Wih_bf, __bf16* __restrict__ Whh_bf,
    int* __restrict__ bar)
{
    int i = blockIdx.x * 256 + threadIdx.x;
    if (i < 4) bar[i] = 0;               // {cnt,gen} x 2 directions
    const int N = G4 * HID;              // 1048576 per matrix
    if (i < N) {
        Wih_bf[i]     = (__bf16)Wihf[i];
        Wih_bf[N + i] = (__bf16)Wihb[i];
        Whh_bf[i]     = (__bf16)Whhf[i];
        Whh_bf[N + i] = (__bf16)Whhb[i];
    }
}

// ---------------------------------------------------------------------------
// Phase 1: x_proj[dir, t*B+b, 4H] = x @ W_ih^T + (b_ih + b_hh)
// Block = 128 threads (4 waves); each wave owns a 16x64 output tile.
// The WG's 64-row W_ih block (contiguous 64KB) is async-staged into LDS and
// shared by all 4 waves.
// ---------------------------------------------------------------------------
__global__ __launch_bounds__(128) void xproj_gemm_kernel(
    const float* __restrict__ x,        // [T*B, I] fp32
    const __bf16* __restrict__ Wih_bf,  // [2][4H, I] bf16
    const float* __restrict__ bihf, const float* __restrict__ bhhf,
    const float* __restrict__ bihb, const float* __restrict__ bhhb,
    float* __restrict__ xproj)          // [2][T*B, 4H] fp32
{
    extern __shared__ char smem[];
    __bf16* sW = (__bf16*)smem;         // [64 rows][512]

    const int nb  = blockIdx.x;   // 0..31  : 64-wide N block
    const int mb  = blockIdx.y;   // 0..511 : 64-wide M block
    const int dir = blockIdx.z;

    const int tid  = threadIdx.x;
    const int wave = tid >> 5;
    const int lane = tid & 31;
    const int lm   = lane & 15;   // M (A) / N (B,C) within tile
    const int lh   = lane >> 4;   // lane half -> K split

    // ---- async-stage the contiguous 64-row W_ih block into LDS ----
    const __bf16* W = Wih_bf + (size_t)dir * (G4 * HID) + (size_t)(nb * 64) * HID;
    for (int i = tid; i < 4096; i += 128)            // 4096 x 16B = 64KB
        async_copy_b128((unsigned)i * 16u, W + (size_t)i * 8);
    async_wait0();
    __syncthreads();

    const float* bih = dir ? bihb : bihf;
    const float* bhh = dir ? bhhb : bhhf;

    const int mrow = mb * 64 + wave * 16;

    v8f acc[4];
#pragma unroll
    for (int nt = 0; nt < 4; ++nt) {
        int n = nb * 64 + nt * 16 + lm;
        float bs = bih[n] + bhh[n];
#pragma unroll
        for (int r = 0; r < 8; ++r) acc[nt][r] = bs;
    }

    const float* xrow = x + (size_t)(mrow + lm) * HID;

#pragma unroll 4
    for (int kc = 0; kc < 16; ++kc) {
        // A fragment (contiguous K: lane half 0 -> K {0..15}, half 1 -> K {16..31})
        const int k0 = kc * 32 + lh * 16;
        float4 f0 = *(const float4*)(xrow + k0);
        float4 f1 = *(const float4*)(xrow + k0 + 4);
        float4 f2 = *(const float4*)(xrow + k0 + 8);
        float4 f3 = *(const float4*)(xrow + k0 + 12);
        v16bf a;
        a[0]=(__bf16)f0.x;  a[1]=(__bf16)f0.y;  a[2]=(__bf16)f0.z;  a[3]=(__bf16)f0.w;
        a[4]=(__bf16)f1.x;  a[5]=(__bf16)f1.y;  a[6]=(__bf16)f1.z;  a[7]=(__bf16)f1.w;
        a[8]=(__bf16)f2.x;  a[9]=(__bf16)f2.y;  a[10]=(__bf16)f2.z; a[11]=(__bf16)f2.w;
        a[12]=(__bf16)f3.x; a[13]=(__bf16)f3.y; a[14]=(__bf16)f3.z; a[15]=(__bf16)f3.w;
#pragma unroll
        for (int nt = 0; nt < 4; ++nt) {
            // B fragment from LDS: column n of B = local row (nt*16+lm) of W_ih
            const __bf16* wr = sW + (nt * 16 + lm) * HID + kc * 32 + lh * 16;
            v16bf bm = *(const v16bf*)wr;
            acc[nt] = __builtin_amdgcn_wmma_f32_16x16x32_bf16(
                false, a, false, bm, (short)0, acc[nt], false, false);
        }
    }

    float* op = xproj + (size_t)dir * ((size_t)T_STEPS * BATCH * G4);
#pragma unroll
    for (int nt = 0; nt < 4; ++nt) {
        int n = nb * 64 + nt * 16 + lm;
#pragma unroll
        for (int r = 0; r < 8; ++r) {
            int m = mrow + r + 8 * lh;            // C/D layout: VGPR r -> M=r / r+8
            op[(size_t)m * G4 + n] = acc[nt][r];
        }
    }
}

// ---------------------------------------------------------------------------
// Grid-wide barrier (per direction, 32 WGs), sense-reversal on L2 atomics.
// ---------------------------------------------------------------------------
static __device__ __forceinline__ void grid_barrier(int* cnt, int* gen, int members) {
    __threadfence();
    __syncthreads();
    if (threadIdx.x == 0) {
        int g = __atomic_load_n(gen, __ATOMIC_RELAXED);
        int prev = __atomic_fetch_add(cnt, 1, __ATOMIC_ACQ_REL);
        if (prev == members - 1) {
            __atomic_store_n(cnt, 0, __ATOMIC_RELAXED);
            __atomic_fetch_add(gen, 1, __ATOMIC_ACQ_REL);   // release new phase
        } else {
            while (__atomic_load_n(gen, __ATOMIC_ACQUIRE) == g)
                __builtin_amdgcn_s_sleep(1);
        }
    }
    __syncthreads();
    __threadfence();
}

// ---------------------------------------------------------------------------
// Phase 2: persistent recurrent kernel. 64 WGs = 2 dirs x 32 h-column slices.
// ---------------------------------------------------------------------------
__global__ __launch_bounds__(128) void lstm_recurrent_kernel(
    const float* __restrict__ xproj,    // [2][T, B, 4H] fp32
    const __bf16* __restrict__ Whh_bf,  // [2][4H, H] bf16
    const float* __restrict__ h0f, const float* __restrict__ c0f,
    const float* __restrict__ h0b, const float* __restrict__ c0b,
    __bf16* __restrict__ hbuf,          // [2][2][B, H] bf16 ping-pong
    float* __restrict__ out,            // [T, B, 2H] ++ hT_f,cT_f,hT_b,cT_b
    int* __restrict__ bar)
{
    extern __shared__ char smem[];
    __bf16* sW = (__bf16*)smem;         // [64 rows][512] : 4 gates x 16 cols

    const int dir   = blockIdx.x >> 5;
    const int slice = blockIdx.x & 31;
    const int cbase = slice * 16;

    const int tid  = threadIdx.x;
    const int wave = tid >> 5;          // M-tile (batch rows)
    const int lane = tid & 31;
    const int lm   = lane & 15;
    const int lh   = lane >> 4;
    const int bbase = wave * 16;

    // ---- async-stage this WG's 64 W_hh rows (4 gates x 16 cols) into LDS ---
    const __bf16* Wd = Whh_bf + (size_t)dir * (G4 * HID);
    for (int i = tid; i < 4096; i += 128) {          // 4096 x 16B = 64KB
        int row = i >> 6;                            // 0..63
        int col = i & 63;                            // 16B chunk within row
        int g = row >> 4, r = row & 15;
        async_copy_b128((unsigned)i * 16u,
                        Wd + (size_t)(g * HID + cbase + r) * HID + col * 8);
    }
    async_wait0();

    // ---- c-state lives in registers in WMMA C/D layout ----
    const float* c0 = dir ? c0b : c0f;
    const float* h0 = dir ? h0b : h0f;
    float creg[8], hlast[8];
#pragma unroll
    for (int r = 0; r < 8; ++r) {
        int b = bbase + r + 8 * lh;
        creg[r]  = c0[b * HID + cbase + lm];
        hlast[r] = 0.0f;
    }

    // ---- publish h0 slice (bf16) into ping buffer 0 ----
    __bf16* hb0 = hbuf + ((size_t)dir * 2 + 0) * (BATCH * HID);
    for (int i = tid; i < 1024; i += 128) {
        int b = i >> 4, c = i & 15;
        hb0[b * HID + cbase + c] = (__bf16)h0[b * HID + cbase + c];
    }
    grid_barrier(&bar[dir * 2], &bar[dir * 2 + 1], 32);

    const float* xpd = xproj + (size_t)dir * ((size_t)T_STEPS * BATCH * G4);

    for (int s = 0; s < T_STEPS; ++s) {
        const int t = dir ? (T_STEPS - 1 - s) : s;
        const __bf16* cur = hbuf + ((size_t)dir * 2 + (s & 1))       * (BATCH * HID);
        __bf16*       nxt = hbuf + ((size_t)dir * 2 + ((s + 1) & 1)) * (BATCH * HID);
        const float*  xp  = xpd + (size_t)t * (BATCH * G4);

        // accumulators start at the precomputed input projection
        v8f acc[4];
#pragma unroll
        for (int g = 0; g < 4; ++g) {
#pragma unroll
            for (int r = 0; r < 8; ++r) {
                int b = bbase + r + 8 * lh;
                acc[g][r] = xp[b * G4 + g * HID + cbase + lm];
            }
        }

        // gates += h @ W_hh^T over K = 512 (16 chunks of 32)
        const __bf16* arow = cur + (bbase + lm) * HID;
#pragma unroll 4
        for (int kc = 0; kc < 16; ++kc) {
            const int k0 = kc * 32 + lh * 16;
            v8bf alo = *(const v8bf*)(arow + k0);
            v8bf ahi = *(const v8bf*)(arow + k0 + 8);
            v16bf a = cat8(alo, ahi);
#pragma unroll
            for (int g = 0; g < 4; ++g) {
                const __bf16* wr = sW + (g * 16 + lm) * HID + kc * 32 + lh * 16;
                v16bf bm = *(const v16bf*)wr;
                acc[g] = __builtin_amdgcn_wmma_f32_16x16x32_bf16(
                    false, a, false, bm, (short)0, acc[g], false, false);
            }
        }

        // LSTM cell update (fp32), publish h_new
#pragma unroll
        for (int r = 0; r < 8; ++r) {
            float ig = sigm(acc[0][r]);
            float fg = sigm(acc[1][r]);
            float gg = tanhf(acc[2][r]);
            float og = sigm(acc[3][r]);
            float cn = fg * creg[r] + ig * gg;
            float hn = og * tanhf(cn);
            creg[r]  = cn;
            hlast[r] = hn;
            int b = bbase + r + 8 * lh;
            nxt[b * HID + cbase + lm] = (__bf16)hn;
            out[(size_t)t * (BATCH * 2 * HID) + (size_t)b * (2 * HID)
                + dir * HID + cbase + lm] = hn;
        }

        grid_barrier(&bar[dir * 2], &bar[dir * 2 + 1], 32);
    }

    // final states: out ++ [hT_f, cT_f, hT_b, cT_b]
    float* tail = out + (size_t)T_STEPS * BATCH * (2 * HID);
#pragma unroll
    for (int r = 0; r < 8; ++r) {
        int b = bbase + r + 8 * lh;
        tail[dir * 2 * BATCH * HID + b * HID + cbase + lm]               = hlast[r];
        tail[dir * 2 * BATCH * HID + BATCH * HID + b * HID + cbase + lm] = creg[r];
    }
}

// ---------------------------------------------------------------------------
extern "C" void kernel_launch(void* const* d_in, const int* in_sizes, int n_in,
                              void* d_out, int out_size, void* d_ws, size_t ws_size,
                              hipStream_t stream) {
    (void)in_sizes; (void)n_in; (void)out_size; (void)ws_size;

    const float* x    = (const float*)d_in[0];
    const float* h0f  = (const float*)d_in[1];
    const float* c0f  = (const float*)d_in[2];
    const float* h0b  = (const float*)d_in[3];
    const float* c0b  = (const float*)d_in[4];
    const float* Wihf = (const float*)d_in[5];
    const float* Whhf = (const float*)d_in[6];
    const float* bihf = (const float*)d_in[7];
    const float* bhhf = (const float*)d_in[8];
    const float* Wihb = (const float*)d_in[9];
    const float* Whhb = (const float*)d_in[10];
    const float* bihb = (const float*)d_in[11];
    const float* bhhb = (const float*)d_in[12];
    float* out = (float*)d_out;

    char* ws = (char*)d_ws;
    int*    bar    = (int*)ws;                                       // 16 B
    __bf16* Whh_bf = (__bf16*)(ws + 256);                            // 4 MiB
    __bf16* Wih_bf = (__bf16*)(ws + 256 + (4u << 20));               // 4 MiB
    __bf16* hbuf   = (__bf16*)(ws + 256 + (8u << 20));               // 256 KiB
    float*  xproj  = (float*) (ws + 256 + (8u << 20) + (256u << 10)); // 512 MiB

    cvt_weights_kernel<<<4096, 256, 0, stream>>>(Wihf, Whhf, Wihb, Whhb,
                                                 Wih_bf, Whh_bf, bar);
    xproj_gemm_kernel<<<dim3(32, 512, 2), 128, 64 * 1024, stream>>>(
        x, Wih_bf, bihf, bhhf, bihb, bhhb, xproj);
    lstm_recurrent_kernel<<<64, 128, 64 * 1024, stream>>>(
        xproj, Whh_bf, h0f, c0f, h0b, c0b, hbuf, out, bar);
}
